// Head_2963527434364
// MI455X (gfx1250) — matrix operand order repaired and checked by reading, not compiled, not run on hardware
//
#include <hip/hip_runtime.h>

#define HEAD 128
#define EMBD 1024
#define BATCH 4
#define SEQ 4096

typedef __attribute__((ext_vector_type(16))) _Float16 v16h;
typedef __attribute__((ext_vector_type(8)))  _Float16 v8h;
typedef __attribute__((ext_vector_type(8)))  float    v8f;
typedef __attribute__((ext_vector_type(4)))  float    v4f;
typedef __attribute__((ext_vector_type(4)))  unsigned int u32x4;
typedef __attribute__((ext_vector_type(8)))  int      i32x8;
typedef __attribute__((ext_vector_type(4)))  int      i32x4;

union AFrag { v16h v; v8h h[2]; };

__device__ __forceinline__ v8f wmma16x16x32(v16h a, v16h b, v8f c) {
  // D = A(16x32 f16) * B(32x16 f16) + C(16x16 f32)
  return __builtin_amdgcn_wmma_f32_16x16x32_f16(
      /*neg_a=*/false, a, /*neg_b=*/false, b,
      /*c_mod=*/(short)0, c, /*reuse_a=*/false, /*reuse_b=*/false);
}

// ---------------------------------------------------------------------------
// TDM: async 2D tile load Global -> LDS (f16 elements), per ISA 08 D# layout.
// group0: [1:0]=count=1, [63:32]=lds_addr, [120:64]=global_addr, [127:126]=2
// group1: [17:16]=data_size(1=2B), tensor_dim0/1, tile_dim0/1, dim0 stride
// ---------------------------------------------------------------------------
__device__ __forceinline__ void tdm_load_2d_f16(
    unsigned lds_off, const _Float16* gptr,
    unsigned tensor_d0, unsigned tensor_d1, unsigned stride0,
    unsigned tile_d0, unsigned tile_d1) {
  unsigned long long ga = (unsigned long long)(size_t)gptr;
  u32x4 g0;
  g0[0] = 1u;                                                  // count=1
  g0[1] = lds_off;                                             // LDS byte addr
  g0[2] = (unsigned)(ga & 0xFFFFFFFFull);                      // ga[31:0]
  g0[3] = (unsigned)((ga >> 32) & 0x01FFFFFFull) | (2u << 30); // ga[56:32]|type=2
  i32x8 g1;
  g1[0] = (int)(1u << 16);                                     // data_size=2B
  g1[1] = (int)(tensor_d0 << 16);                              // td0[15:0]
  g1[2] = (int)((tensor_d0 >> 16) | (tensor_d1 << 16));        // td0 hi | td1 lo
  g1[3] = (int)((tensor_d1 >> 16) | (tile_d0 << 16));          // td1 hi | tile0
  g1[4] = (int)(tile_d1 & 0xFFFFu);                            // tile1, tile2=0
  g1[5] = (int)stride0;                                        // stride0[31:0]
  g1[6] = 0;                                                   // stride0 hi
  g1[7] = 0;
  i32x4 gz = {0, 0, 0, 0};
#if __clang_major__ >= 23
  i32x8 gz8 = {0, 0, 0, 0, 0, 0, 0, 0};
  __builtin_amdgcn_tensor_load_to_lds(g0, g1, gz, gz, gz8, 0);
#else
  __builtin_amdgcn_tensor_load_to_lds(g0, g1, gz, gz, 0);
#endif
}

// ---------------------------------------------------------------------------
// Kernel 0: convert + transpose weights -> Wt f16 [3][HEAD][EMBD]
// ---------------------------------------------------------------------------
__global__ void wcvt_kernel(const float* __restrict__ Wq,
                            const float* __restrict__ Wk,
                            const float* __restrict__ Wv,
                            _Float16* __restrict__ Wt) {
  int idx = blockIdx.x * blockDim.x + threadIdx.x;   // 3*HEAD*EMBD threads
  int w   = idx / (HEAD * EMBD);
  int rem = idx % (HEAD * EMBD);
  int n   = rem / EMBD;     // head col
  int c   = rem % EMBD;     // emb row
  const float* W = (w == 0) ? Wq : (w == 1) ? Wk : Wv;
  Wt[idx] = (_Float16)W[(size_t)c * HEAD + n];
}

// ---------------------------------------------------------------------------
// Kernel 1: QKV projection. One wave computes a 16x128 strip of X @ W.
// grid = (B*T/16, 3), block = 32 (one wave).
// Outputs: Qh, Kh as f16 [B*T][HEAD]; V stored transposed Vt f16 [B][HEAD][T].
// ---------------------------------------------------------------------------
__global__ __launch_bounds__(32) void qkv_wmma_kernel(
    const float* __restrict__ x, const _Float16* __restrict__ Wt,
    _Float16* __restrict__ Qh, _Float16* __restrict__ Kh,
    _Float16* __restrict__ Vt) {
  const int lane = threadIdx.x;
  const int half = lane >> 4;
  const int lm   = lane & 15;
  const int rowbase = blockIdx.x * 16;
  const int w       = blockIdx.y;
  const _Float16* Ww = Wt + (size_t)w * HEAD * EMBD;

  v8f acc[8] = {};
  const float* xrow = x + (size_t)(rowbase + lm) * EMBD;

  for (int kb = 0; kb < EMBD; kb += 32) {
    // A fragment: 16x32 chunk of x, converted f32->f16.
    // lane (half h) holds row lm, K = kb + {0..7,16..23} + 8*h
    AFrag a;
    const float* xs = xrow + kb + 8 * half;
    v4f x0 = *(const v4f*)(xs + 0);
    v4f x1 = *(const v4f*)(xs + 4);
    v4f x2 = *(const v4f*)(xs + 16);
    v4f x3 = *(const v4f*)(xs + 20);
#pragma unroll
    for (int e = 0; e < 4; ++e) {
      a.v[e]      = (_Float16)x0[e];
      a.v[4 + e]  = (_Float16)x1[e];
      a.v[8 + e]  = (_Float16)x2[e];
      a.v[12 + e] = (_Float16)x3[e];
    }
#pragma unroll
    for (int nt = 0; nt < 8; ++nt) {
      // B fragment: lane holds col n = nt*16+lm, K = kb + {0..7,16..23} + 8*half
      const _Float16* wr = Ww + (size_t)(nt * 16 + lm) * EMBD + kb + 8 * half;
      AFrag bf;
      bf.h[0] = *(const v8h*)(wr);
      bf.h[1] = *(const v8h*)(wr + 16);
      acc[nt] = wmma16x16x32(a.v, bf.v, acc[nt]);
    }
  }

  // Epilogue: D tile VGPR r <-> row rowbase + r + 8*half, col nt*16 + lm
#pragma unroll
  for (int nt = 0; nt < 8; ++nt) {
    const int col = nt * 16 + lm;
#pragma unroll
    for (int r = 0; r < 8; ++r) {
      const int row = rowbase + r + 8 * half;
      const _Float16 vle = (_Float16)acc[nt][r];
      if (w == 0) {
        Qh[(size_t)row * HEAD + col] = vle;
      } else if (w == 1) {
        Kh[(size_t)row * HEAD + col] = vle;
      } else {
        const int bb = row >> 12;          // row / SEQ
        const int tt = row & (SEQ - 1);
        Vt[((size_t)(bb * HEAD + col)) * SEQ + tt] = vle;
      }
    }
  }
}

// ---------------------------------------------------------------------------
// Kernel 2: causal flash attention. One wave per 16-query tile.
// K/V tiles are DMA'd into LDS by the Tensor Data Mover, double-buffered,
// synchronized with TENSORcnt. grid = (T/16, B), block = 32.
// ---------------------------------------------------------------------------
__global__ __launch_bounds__(32) void attn_wmma_kernel(
    const _Float16* __restrict__ Qh, const _Float16* __restrict__ Kh,
    const _Float16* __restrict__ Vt, float* __restrict__ out) {
  __shared__ __align__(16) _Float16 Klds[2][32 * HEAD];   // [key][head]
  __shared__ __align__(16) _Float16 Vlds[2][HEAD * 32];   // [head][key]
  __shared__ __align__(32) _Float16 Plds[16 * 32];

  const int lane = threadIdx.x;
  const int half = lane >> 4;
  const int lm   = lane & 15;
  const int qt   = blockIdx.x;
  const int b    = blockIdx.y;
  const int rowbase = qt * 16;

  const _Float16* Qb = Qh + ((size_t)b * SEQ + rowbase) * HEAD;
  const _Float16* Kb = Kh + (size_t)b * SEQ * HEAD;
  const _Float16* Vb = Vt + (size_t)b * HEAD * SEQ;

  // Q fragments: 16x128 as four 16x32 A-fragments, resident for whole loop.
  AFrag qa[4];
#pragma unroll
  for (int kk = 0; kk < 4; ++kk) {
    const _Float16* qs = Qb + lm * HEAD + kk * 32 + 8 * half;
    qa[kk].h[0] = *(const v8h*)(qs);
    qa[kk].h[1] = *(const v8h*)(qs + 16);
  }

  v8f acc[8] = {};
  float mrun[8], lrun[8];
#pragma unroll
  for (int r = 0; r < 8; ++r) { mrun[r] = -1e30f; lrun[r] = 0.f; }

  const float scale = 0.08838834764831845f;  // 1/sqrt(128)
  const int nblocks = (rowbase + 47) >> 5;   // causal: keys <= rowbase+15

  // ---- TDM prefetch of block 0 (K: 32x128 contiguous; V: 128 rows of 32)
  tdm_load_2d_f16((unsigned)(size_t)&Klds[0][0], Kb,
                  /*td0=*/HEAD, /*td1=*/SEQ, /*stride0=*/HEAD,
                  /*tile0=*/HEAD, /*tile1=*/32);
  tdm_load_2d_f16((unsigned)(size_t)&Vlds[0][0], Vb,
                  /*td0=*/SEQ, /*td1=*/HEAD, /*stride0=*/SEQ,
                  /*tile0=*/32, /*tile1=*/HEAD);

  for (int blk = 0; blk < nblocks; ++blk) {
    const int s0  = blk * 32;
    const int buf = blk & 1;
    const int nb  = blk + 1;

    if (nb < nblocks) {   // prefetch next block into the other buffer
      const int nbuf = nb & 1;
      tdm_load_2d_f16((unsigned)(size_t)&Klds[nbuf][0],
                      Kb + (size_t)(nb * 32) * HEAD,
                      HEAD, SEQ, HEAD, HEAD, 32);
      tdm_load_2d_f16((unsigned)(size_t)&Vlds[nbuf][0],
                      Vb + (size_t)(nb * 32),
                      SEQ, HEAD, SEQ, 32, HEAD);
      __builtin_amdgcn_s_wait_tensorcnt(2);  // current block's 2 loads done
    } else {
      __builtin_amdgcn_s_wait_tensorcnt(0);
    }

    // ---- S = Q @ K^T : two 16x16 tiles over a 32-key block (from LDS)
    v8f st[2] = {};
#pragma unroll
    for (int j = 0; j < 2; ++j) {
      const _Float16* kr = &Klds[buf][(16 * j + lm) * HEAD + 8 * half];
#pragma unroll
      for (int kk = 0; kk < 4; ++kk) {
        AFrag bf;
        bf.h[0] = *(const v8h*)(kr + kk * 32);
        bf.h[1] = *(const v8h*)(kr + kk * 32 + 16);
        st[j] = wmma16x16x32(qa[kk].v, bf.v, st[j]);
      }
    }

    // ---- scale, causal mask, online softmax
    float alpha[8];
#pragma unroll
    for (int r = 0; r < 8; ++r) {
      const int row = rowbase + r + 8 * half;
      float s0v = st[0][r] * scale;
      float s1v = st[1][r] * scale;
      if (s0 + lm > row)      s0v = -1e30f;
      if (s0 + 16 + lm > row) s1v = -1e30f;
      float rm = fmaxf(s0v, s1v);
#pragma unroll
      for (int m = 8; m >= 1; m >>= 1) rm = fmaxf(rm, __shfl_xor(rm, m, 32));
      const float mnew = fmaxf(mrun[r], rm);
      alpha[r] = __expf(mrun[r] - mnew);
      mrun[r]  = mnew;
      const float p0 = __expf(s0v - mnew);
      const float p1 = __expf(s1v - mnew);
      Plds[(r + 8 * half) * 32 + lm]      = (_Float16)p0;
      Plds[(r + 8 * half) * 32 + 16 + lm] = (_Float16)p1;
      float ps = p0 + p1;
#pragma unroll
      for (int m = 8; m >= 1; m >>= 1) ps += __shfl_xor(ps, m, 32);
      lrun[r] = lrun[r] * alpha[r] + ps;
    }

    // ---- rescale running output
#pragma unroll
    for (int nt = 0; nt < 8; ++nt)
#pragma unroll
      for (int r = 0; r < 8; ++r) acc[nt][r] *= alpha[r];

    // LDS round-trip transposes P from C/D layout into A layout.
    asm volatile("s_wait_dscnt 0" ::: "memory");

    AFrag pa;
    {
      const _Float16* ps = &Plds[lm * 32 + 8 * half];
      pa.h[0] = *(const v8h*)(ps);
      pa.h[1] = *(const v8h*)(ps + 16);
    }

    // ---- O += P @ V (V tile in LDS, key index contiguous per head row)
#pragma unroll
    for (int nt = 0; nt < 8; ++nt) {
      const _Float16* vr = &Vlds[buf][(nt * 16 + lm) * 32 + 8 * half];
      AFrag vf;
      vf.h[0] = *(const v8h*)(vr);
      vf.h[1] = *(const v8h*)(vr + 16);
      acc[nt] = wmma16x16x32(pa.v, vf.v, acc[nt]);
    }
  }

  // ---- epilogue: divide by row sums, store f32
  float* ob = out + ((size_t)b * SEQ + rowbase) * HEAD;
#pragma unroll
  for (int nt = 0; nt < 8; ++nt) {
    const int col = nt * 16 + lm;
#pragma unroll
    for (int r = 0; r < 8; ++r) {
      ob[(size_t)(r + 8 * half) * HEAD + col] = acc[nt][r] / lrun[r];
    }
  }
}

// ---------------------------------------------------------------------------
extern "C" void kernel_launch(void* const* d_in, const int* in_sizes, int n_in,
                              void* d_out, int out_size, void* d_ws, size_t ws_size,
                              hipStream_t stream) {
  const float* x  = (const float*)d_in[0];
  const float* Wq = (const float*)d_in[1];
  const float* Wk = (const float*)d_in[2];
  const float* Wv = (const float*)d_in[3];
  float* out = (float*)d_out;

  // Workspace layout (f16): Q [B*T][H], K [B*T][H], Vt [B][H][T], Wt [3][H][C]
  const size_t elems = (size_t)BATCH * SEQ * HEAD;   // 2,097,152
  _Float16* Qh = (_Float16*)d_ws;
  _Float16* Kh = Qh + elems;
  _Float16* Vt = Kh + elems;
  _Float16* Wt = Vt + elems;

  {
    const int total = 3 * HEAD * EMBD;
    wcvt_kernel<<<dim3((total + 255) / 256), dim3(256), 0, stream>>>(Wq, Wk, Wv, Wt);
  }
  {
    dim3 g(BATCH * SEQ / 16, 3);
    qkv_wmma_kernel<<<g, dim3(32), 0, stream>>>(x, Wt, Qh, Kh, Vt);
  }
  {
    dim3 g(SEQ / 16, BATCH);
    attn_wmma_kernel<<<g, dim3(32), 0, stream>>>(Qh, Kh, Vt, out);
  }
}
